// Variation_3066606649766
// MI455X (gfx1250) — matrix operand order repaired and checked by reading, not compiled, not run on hardware
//
#include <hip/hip_runtime.h>

// ---------------------------------------------------------------------------
// Problem dims (compile-time)
// ---------------------------------------------------------------------------
#define MDIM   8192      // B*S
#define DIN    4096
#define ZDIM   1024
#define DOUT   4096
#define CATK   5120      // Z + D_IN
#define MU_OFF  33554432u            // MDIM*DOUT
#define STD_OFF 41943040u            // MU_OFF + MDIM*ZDIM

// GEMM tiling
#define BM  128
#define BN  128
#define BK  32
#define LDT 40            // LDS row stride in bf16 elems (BK + 8 pad, 80B = 5*16B)

typedef __attribute__((ext_vector_type(16))) __bf16 v16bf;
typedef __attribute__((ext_vector_type(8)))  __bf16 v8bf;
typedef __attribute__((ext_vector_type(8)))  float  v8f;

// ---------------------------------------------------------------------------
// f32 -> bf16 (round-to-nearest-even) helpers
// ---------------------------------------------------------------------------
__device__ __forceinline__ unsigned f2bf_u(float f) {
    unsigned u = __float_as_uint(f);
    return (u + 0x7FFFu + ((u >> 16) & 1u)) >> 16;
}
__device__ __forceinline__ unsigned pk2(float a, float b) {
    return f2bf_u(a) | (f2bf_u(b) << 16);
}
__device__ __forceinline__ unsigned short f2bf(float f) {
    return (unsigned short)f2bf_u(f);
}

// ---------------------------------------------------------------------------
// WMMA fragment loads from LDS (ISA 7.12.2 layouts, wave32)
// A: lane holds row (lane&15); K = 8*(lane>>4) + {0..7, 16..23}
// B: lane holds col (lane&15); K = 8*(lane>>4) + {0..7, 16..23}
// ---------------------------------------------------------------------------
__device__ __forceinline__ v16bf ldfragA(const unsigned short* l, int row, int lane) {
    const unsigned short* p = l + row * LDT + ((lane >> 4) << 3);
    v8bf lo = *(const v8bf*)p;               // ds_load_b128
    v8bf hi = *(const v8bf*)(p + 16);        // ds_load_b128
    return __builtin_shufflevector(lo, hi, 0,1,2,3,4,5,6,7,8,9,10,11,12,13,14,15);
}
__device__ __forceinline__ v16bf ldfragB(const unsigned short* l, int row, int lane) {
    const unsigned short* p = l + row * LDT + ((lane >> 4) << 3);
    v8bf lo = *(const v8bf*)p;
    v8bf hi = *(const v8bf*)(p + 16);
    return __builtin_shufflevector(lo, hi, 0,1,2,3,4,5,6,7,8,9,10,11,12,13,14,15);
}

// 2x4 tile of v_wmma_f32_16x16x32_bf16 on one LDS stage
__device__ __forceinline__ void compute_tile(v8f (&acc)[2][4],
                                             const unsigned short* lA,
                                             const unsigned short* lB,
                                             int wm, int wn, int lane) {
    const int lrow = lane & 15;
    v16bf af[2], bfr[4];
#pragma unroll
    for (int i = 0; i < 2; ++i) af[i]  = ldfragA(lA, wm + i * 16 + lrow, lane);
#pragma unroll
    for (int j = 0; j < 4; ++j) bfr[j] = ldfragB(lB, wn + j * 16 + lrow, lane);
#pragma unroll
    for (int i = 0; i < 2; ++i)
#pragma unroll
        for (int j = 0; j < 4; ++j)
            acc[i][j] = __builtin_amdgcn_wmma_f32_16x16x32_bf16(
                false, af[i], false, bfr[j], (short)0, acc[i][j], false, false);
}

// ---------------------------------------------------------------------------
// GEMM: C[m,n] = sum_k A[m,k] * W[n,k] + bias[n]
//   A: [M,K] bf16, W: [N,K] bf16, C: [M,N] f32, LDC compile-time
//   grid = (N/BN, M/BM), block = 256 (8 waves)
//   Double-buffered LDS, one barrier per K-step; global loads for stage k+1
//   and ds_stores to the idle buffer overlap the WMMAs on stage k.
// ---------------------------------------------------------------------------
template <int LDC>
__global__ __launch_bounds__(256)
void k_gemm(const unsigned short* __restrict__ A, int lda,
            const unsigned short* __restrict__ W, int ldb,
            const float* __restrict__ bias,
            float* __restrict__ C, int K)
{
    __shared__ __align__(16) unsigned short lA[2][BM * LDT];
    __shared__ __align__(16) unsigned short lB[2][BN * LDT];

    const int tid  = threadIdx.x;
    const int lane = tid & 31;
    const int wid  = tid >> 5;
    const int bm   = blockIdx.y * BM;
    const int bn   = blockIdx.x * BN;
    const int wm   = (wid & 3) * 32;    // wave sub-tile: 32 rows
    const int wn   = (wid >> 2) * 64;   //                64 cols
    const int lrow = lane & 15;
    const int lhalf = lane >> 4;

    // staging assignment: thread -> (row, 16-elem half of BK)
    const int srow = tid >> 1;
    const int scol = (tid & 1) << 4;
    const int soff = srow * LDT + scol;

    const unsigned short* pa = A + (size_t)(bm + srow) * lda + scol;
    const unsigned short* pb = W + (size_t)(bn + srow) * ldb + scol;

    // prologue: stage tile 0 into buffer 0
    {
        uint4 ra0 = ((const uint4*)pa)[0], ra1 = ((const uint4*)pa)[1];
        uint4 rb0 = ((const uint4*)pb)[0], rb1 = ((const uint4*)pb)[1];
        *(uint4*)&lA[0][soff] = ra0; *(uint4*)&lA[0][soff + 8] = ra1;
        *(uint4*)&lB[0][soff] = rb0; *(uint4*)&lB[0][soff + 8] = rb1;
    }
    __syncthreads();

    v8f acc[2][4] = {};
    int cur = 0;

    for (int k0 = BK; k0 < K; k0 += BK) {
        // fetch next tile (overlaps with WMMA below)
        const unsigned short* qa = pa + k0;
        const unsigned short* qb = pb + k0;
        uint4 ra0 = ((const uint4*)qa)[0], ra1 = ((const uint4*)qa)[1];
        uint4 rb0 = ((const uint4*)qb)[0], rb1 = ((const uint4*)qb)[1];
        __builtin_prefetch(qa + BK, 0, 3);     // global_prefetch_b8, 2 tiles ahead
        __builtin_prefetch(qb + BK, 0, 3);

        compute_tile(acc, lA[cur], lB[cur], wm, wn, lane);

        const int nxt = cur ^ 1;               // stage into idle buffer
        *(uint4*)&lA[nxt][soff] = ra0; *(uint4*)&lA[nxt][soff + 8] = ra1;
        *(uint4*)&lB[nxt][soff] = rb0; *(uint4*)&lB[nxt][soff + 8] = rb1;

        __syncthreads();                       // publish nxt, retire reads of cur
        cur = nxt;
    }
    compute_tile(acc, lA[cur], lB[cur], wm, wn, lane);   // last tile

    // Epilogue: +bias, f32 store. C/D layout: m = 16i + 8*lhalf + v, n = lrow.
#pragma unroll
    for (int j = 0; j < 4; ++j) {
        const int n = bn + wn + j * 16 + lrow;
        const float bv = bias[n];
#pragma unroll
        for (int i = 0; i < 2; ++i) {
            float* cp = C + (size_t)(bm + wm + i * 16 + lhalf * 8) * LDC + n;
#pragma unroll
            for (int v = 0; v < 8; ++v)
                cp[(size_t)v * LDC] = acc[i][j][v] + bv;   // constant offsets
        }
    }
}

// ---------------------------------------------------------------------------
// f32 -> bf16 bulk conversion (4 elements/thread); count = grid*1024
// ---------------------------------------------------------------------------
__global__ __launch_bounds__(256)
void k_cvt_bf16(const float* __restrict__ s, unsigned short* __restrict__ d)
{
    const size_t i = ((size_t)blockIdx.x * 256 + threadIdx.x) * 4;
    float4 v = *(const float4*)(s + i);
    uint2 p;
    p.x = pk2(v.x, v.y);
    p.y = pk2(v.z, v.w);
    *(uint2*)(d + i) = p;
}

// ---------------------------------------------------------------------------
// Row LayerNorm(1024) + tanh, f32 -> bf16.  One block per row.
// ---------------------------------------------------------------------------
__global__ __launch_bounds__(256)
void k_ln_tanh(const float* __restrict__ pre,
               const float* __restrict__ g, const float* __restrict__ be,
               unsigned short* __restrict__ out)
{
    const int row = blockIdx.x;
    const int tid = threadIdx.x;
    const float* p = pre + (size_t)row * ZDIM;

    float x0 = p[tid], x1 = p[tid + 256], x2 = p[tid + 512], x3 = p[tid + 768];
    float s = x0 + x1 + x2 + x3;
    float q = x0 * x0 + x1 * x1 + x2 * x2 + x3 * x3;
#pragma unroll
    for (int o = 16; o > 0; o >>= 1) {
        s += __shfl_xor(s, o, 32);
        q += __shfl_xor(q, o, 32);
    }
    __shared__ float rs[8], rq[8];
    if ((tid & 31) == 0) { rs[tid >> 5] = s; rq[tid >> 5] = q; }
    __syncthreads();
    float S = 0.f, Q = 0.f;
#pragma unroll
    for (int w = 0; w < 8; ++w) { S += rs[w]; Q += rq[w]; }

    const float mean = S * (1.f / 1024.f);
    const float var  = Q * (1.f / 1024.f) - mean * mean;
    const float rstd = rsqrtf(var + 1e-5f);

    unsigned short* o0 = out + (size_t)row * ZDIM;
#pragma unroll
    for (int e = 0; e < 4; ++e) {
        const int c = tid + e * 256;
        const float xv = (e == 0 ? x0 : e == 1 ? x1 : e == 2 ? x2 : x3);
        o0[c] = f2bf(tanhf((xv - mean) * rstd * g[c] + be[c]));
    }
}

// ---------------------------------------------------------------------------
// Reparameterization: std = exp(0.5*ls) (in place over ls slot of d_out),
// z = eps*std + mu, concat[:, 0:Z] = bf16(tanh(z))
// ---------------------------------------------------------------------------
__global__ __launch_bounds__(256)
void k_reparam(float* __restrict__ outbuf, const float* __restrict__ eps,
               unsigned short* __restrict__ cat)
{
    const unsigned i = blockIdx.x * 256u + threadIdx.x;   // < MDIM*ZDIM
    const float mu = outbuf[MU_OFF + i];
    const float ls = outbuf[STD_OFF + i];
    const float sd = expf(0.5f * ls);
    outbuf[STD_OFF + i] = sd;
    const float z = eps[i] * sd + mu;
    const unsigned row = i >> 10, col = i & 1023u;
    cat[(size_t)row * CATK + col] = f2bf(tanhf(z));
}

// concat[:, Z:Z+D_IN] = bf16(tanh(x))
__global__ __launch_bounds__(256)
void k_tanhx(const float* __restrict__ x, unsigned short* __restrict__ cat)
{
    const unsigned i = blockIdx.x * 256u + threadIdx.x;   // < MDIM*DIN
    const unsigned row = i >> 12, col = i & 4095u;
    cat[(size_t)row * CATK + ZDIM + col] = f2bf(tanhf(x[i]));
}

// ---------------------------------------------------------------------------
// Launch
// ---------------------------------------------------------------------------
extern "C" void kernel_launch(void* const* d_in, const int* in_sizes, int n_in,
                              void* d_out, int out_size, void* d_ws, size_t ws_size,
                              hipStream_t stream)
{
    const float* x   = (const float*)d_in[0];
    const float* eps = (const float*)d_in[1];
    const float* W1  = (const float*)d_in[2];
    const float* b1  = (const float*)d_in[3];
    const float* g1  = (const float*)d_in[4];
    const float* be1 = (const float*)d_in[5];
    const float* W2  = (const float*)d_in[6];
    const float* b2  = (const float*)d_in[7];
    const float* g2  = (const float*)d_in[8];
    const float* be2 = (const float*)d_in[9];
    const float* Wmu = (const float*)d_in[10];
    const float* bmu = (const float*)d_in[11];
    const float* Wls = (const float*)d_in[12];
    const float* bls = (const float*)d_in[13];
    const float* Wzw = (const float*)d_in[14];
    const float* bzw = (const float*)d_in[15];
    float* out = (float*)d_out;

    // workspace layout (bytes):
    //   pre f32 [M,Z]         @ 0        (32 MB)
    //   h   bf16 [M,Z]        @ 32 MB    (16 MB)
    //   cat bf16 [M,CATK]     @ 48 MB    (80 MB)
    //   x   bf16 [M,DIN]      @ 128 MB   (64 MB)
    //   W1/W2/Wmu/Wls/Wzw bf16 @ 192 MB  (54 MB)
    char* w = (char*)d_ws;
    float*          pre  = (float*)w;
    unsigned short* hbuf = (unsigned short*)(w + (size_t)33554432);
    unsigned short* cat  = (unsigned short*)(w + (size_t)50331648);
    unsigned short* xb   = (unsigned short*)(w + (size_t)134217728);
    unsigned short* W1b  = (unsigned short*)(w + (size_t)201326592);
    unsigned short* W2b  = W1b + (size_t)ZDIM * DIN;
    unsigned short* Wmub = W2b + (size_t)ZDIM * ZDIM;
    unsigned short* Wlsb = Wmub + (size_t)ZDIM * ZDIM;
    unsigned short* Wzwb = Wlsb + (size_t)ZDIM * ZDIM;

    const dim3 blk(256);

    // one-shot bf16 conversions (bandwidth-trivial)
    k_cvt_bf16<<<(MDIM * DIN) / 1024, blk, 0, stream>>>(x, xb);
    k_cvt_bf16<<<(ZDIM * DIN) / 1024, blk, 0, stream>>>(W1, W1b);
    k_cvt_bf16<<<(ZDIM * ZDIM) / 1024, blk, 0, stream>>>(W2, W2b);
    k_cvt_bf16<<<(ZDIM * ZDIM) / 1024, blk, 0, stream>>>(Wmu, Wmub);
    k_cvt_bf16<<<(ZDIM * ZDIM) / 1024, blk, 0, stream>>>(Wls, Wlsb);
    k_cvt_bf16<<<(DOUT * CATK) / 1024, blk, 0, stream>>>(Wzw, Wzwb);

    // concat right half: tanh(x) -> bf16  (independent of the MLP chain)
    k_tanhx<<<(MDIM * DIN) / 256, blk, 0, stream>>>(x, cat);

    // layer 1: pre = x @ W1^T + b1 ; h = tanh(LN(pre))
    k_gemm<ZDIM><<<dim3(ZDIM / BN, MDIM / BM), blk, 0, stream>>>(
        xb, DIN, W1b, DIN, b1, pre, DIN);
    k_ln_tanh<<<MDIM, blk, 0, stream>>>(pre, g1, be1, hbuf);

    // layer 2
    k_gemm<ZDIM><<<dim3(ZDIM / BN, MDIM / BM), blk, 0, stream>>>(
        hbuf, ZDIM, W2b, ZDIM, b2, pre, ZDIM);
    k_ln_tanh<<<MDIM, blk, 0, stream>>>(pre, g2, be2, hbuf);

    // mu and logsigma heads (logsigma parked in d_out's std slot)
    k_gemm<ZDIM><<<dim3(ZDIM / BN, MDIM / BM), blk, 0, stream>>>(
        hbuf, ZDIM, Wmub, ZDIM, bmu, out + MU_OFF, ZDIM);
    k_gemm<ZDIM><<<dim3(ZDIM / BN, MDIM / BM), blk, 0, stream>>>(
        hbuf, ZDIM, Wlsb, ZDIM, bls, out + STD_OFF, ZDIM);

    // std/z/tanh(z) -> concat left half
    k_reparam<<<(MDIM * ZDIM) / 256, blk, 0, stream>>>(out, eps, cat);

    // output head: out = tanh([z|x]) @ Wzw^T + bzw   (dominant GEMM)
    k_gemm<DOUT><<<dim3(DOUT / BN, MDIM / BM), blk, 0, stream>>>(
        cat, CATK, Wzwb, CATK, bzw, out, CATK);
}
